// MultiHeadAttention_12455405158937
// MI455X (gfx1250) — matrix operand order repaired and checked by reading, not compiled, not run on hardware
//
#include <hip/hip_runtime.h>
#include <hip/hip_bf16.h>

typedef _Float16 f16;
typedef __attribute__((ext_vector_type(16))) _Float16 v16h;
typedef __attribute__((ext_vector_type(8)))  _Float16 v8h;
typedef __attribute__((ext_vector_type(8)))  float    v8f;

#define D_MODEL 768
#define SEQ     1024
#define BATCH   8
#define HEADS   12
#define HD      64
#define M_ROWS  (BATCH*SEQ)   /* 8192 */

__device__ __forceinline__ v8f wmma16(v16h a, v16h b, v8f c) {
  // v_wmma_f32_16x16x32_f16: D = A(16x32 f16) x B(32x16 f16) + C(16x16 f32)
  return __builtin_amdgcn_wmma_f32_16x16x32_f16(false, a, false, b, (short)0, c, false, false);
}

__device__ __forceinline__ v16h cat8(v8h lo, v8h hi) {
  return __builtin_shufflevector(lo, hi, 0,1,2,3,4,5,6,7,8,9,10,11,12,13,14,15);
}

// Reductions within each 16-lane half of the wave32 (columns of a C fragment).
__device__ __forceinline__ float halfmax(float v) {
  v = fmaxf(v, __shfl_xor(v, 8, 32));
  v = fmaxf(v, __shfl_xor(v, 4, 32));
  v = fmaxf(v, __shfl_xor(v, 2, 32));
  v = fmaxf(v, __shfl_xor(v, 1, 32));
  return v;
}
__device__ __forceinline__ float halfsum(float v) {
  v += __shfl_xor(v, 8, 32);
  v += __shfl_xor(v, 4, 32);
  v += __shfl_xor(v, 2, 32);
  v += __shfl_xor(v, 1, 32);
  return v;
}

// ---------------- conversion kernels ----------------
__global__ void cvt_f16_kernel(const float* __restrict__ src, f16* __restrict__ dst, int n) {
  int i = blockIdx.x * blockDim.x + threadIdx.x;
  if (i < n) dst[i] = (f16)src[i];
}

// dst[c*rows + r] = (f16)src[r*cols + c]   (weight transpose for contiguous B loads)
__global__ void cvt_f16_t_kernel(const float* __restrict__ src, f16* __restrict__ dst,
                                 int rows, int cols) {
  int i = blockIdx.x * blockDim.x + threadIdx.x;
  if (i < rows * cols) {
    int r = i / cols, c = i % cols;
    dst[(size_t)c * rows + r] = (f16)src[i];
  }
}

// ---------------- register-blocked 32x64 GEMM tile per wave ----------------
// acc[mi][f]: rows row0+16*mi.., cols col0+16*f..; A reused 4x, B reused 2x
// -> 8 WMMAs per k-step from 6 fragment loads.
__device__ __forceinline__ void gemm_tile_32x64(
    const f16* __restrict__ X, const f16* __restrict__ Wt,
    int row0, int col0, int lr, int lh, v8f acc[2][4]) {
  const f16* xr0 = X + (size_t)(row0 + lr) * D_MODEL + lh * 16;
  const f16* xr1 = X + (size_t)(row0 + 16 + lr) * D_MODEL + lh * 16;
  const f16* wr0 = Wt + (size_t)(col0 + lr) * D_MODEL + lh * 16;
  const f16* wr1 = wr0 + (size_t)16 * D_MODEL;
  const f16* wr2 = wr0 + (size_t)32 * D_MODEL;
  const f16* wr3 = wr0 + (size_t)48 * D_MODEL;
#pragma unroll 2
  for (int kt = 0; kt < D_MODEL / 32; ++kt) {
    const int k0 = kt * 32;
    const v16h a0 = cat8(*(const v8h*)(xr0 + k0), *(const v8h*)(xr0 + k0 + 8));
    const v16h a1 = cat8(*(const v8h*)(xr1 + k0), *(const v8h*)(xr1 + k0 + 8));
    const v16h b0 = *(const v16h*)(wr0 + k0);
    const v16h b1 = *(const v16h*)(wr1 + k0);
    const v16h b2 = *(const v16h*)(wr2 + k0);
    const v16h b3 = *(const v16h*)(wr3 + k0);
    acc[0][0] = wmma16(a0, b0, acc[0][0]);
    acc[1][0] = wmma16(a1, b0, acc[1][0]);
    acc[0][1] = wmma16(a0, b1, acc[0][1]);
    acc[1][1] = wmma16(a1, b1, acc[1][1]);
    acc[0][2] = wmma16(a0, b2, acc[0][2]);
    acc[1][2] = wmma16(a1, b2, acc[1][2]);
    acc[0][3] = wmma16(a0, b3, acc[0][3]);
    acc[1][3] = wmma16(a1, b3, acc[1][3]);
  }
}

// ---------------- projection GEMM: Out = X(8192x768) * W + b ----------------
// X: f16 row-major. Wt: f16 [n][k] (transposed). Out: f16, per-head layout.
// transposeOut==0 -> [b,h,n,d]; transposeOut==1 -> [b,h,d,n]
__global__ void __launch_bounds__(256) proj_gemm_kernel(
    const f16* __restrict__ X, const f16* __restrict__ Wt,
    const float* __restrict__ bias, f16* __restrict__ Out, int transposeOut) {
  const int lane = threadIdx.x & 31;
  const int wid  = blockIdx.x * (blockDim.x >> 5) + (threadIdx.x >> 5);
  const int NT   = D_MODEL / 64;              // 12 col tiles
  const int mtile = wid / NT, ntile = wid % NT;
  const int lr = lane & 15, lh = lane >> 4;
  const int row0 = mtile * 32, col0 = ntile * 64;

  v8f acc[2][4] = {};
  gemm_tile_32x64(X, Wt, row0, col0, lr, lh, acc);

#pragma unroll
  for (int f = 0; f < 4; ++f) {
    const int c  = col0 + 16 * f + lr;
    const float bv = bias[c];
    const int hh = c >> 6, dd = c & 63;
#pragma unroll
    for (int mi = 0; mi < 2; ++mi) {
#pragma unroll
      for (int r = 0; r < 8; ++r) {
        const int m  = row0 + 16 * mi + r + lh * 8;
        const int bI = m >> 10, nI = m & 1023;
        const float v = acc[mi][f][r] + bv;
        size_t idx = transposeOut
            ? ((((size_t)bI * HEADS + hh) * HD + dd) * SEQ + nI)
            : ((((size_t)bI * HEADS + hh) * SEQ + nI) * HD + dd);
        Out[idx] = (f16)v;
      }
    }
  }
}

// ---------------- output GEMM: out = Ctx(8192x768) * Wo + bo (fp32 out) -----
__global__ void __launch_bounds__(256) out_gemm_kernel(
    const f16* __restrict__ X, const f16* __restrict__ Wt,
    const float* __restrict__ bias, float* __restrict__ Out) {
  const int lane = threadIdx.x & 31;
  const int wid  = blockIdx.x * (blockDim.x >> 5) + (threadIdx.x >> 5);
  const int NT   = D_MODEL / 64;
  const int mtile = wid / NT, ntile = wid % NT;
  const int lr = lane & 15, lh = lane >> 4;
  const int row0 = mtile * 32, col0 = ntile * 64;

  v8f acc[2][4] = {};
  gemm_tile_32x64(X, Wt, row0, col0, lr, lh, acc);

#pragma unroll
  for (int f = 0; f < 4; ++f) {
    const int c = col0 + 16 * f + lr;
    const float bv = bias[c];
#pragma unroll
    for (int mi = 0; mi < 2; ++mi) {
#pragma unroll
      for (int r = 0; r < 8; ++r) {
        const int m = row0 + 16 * mi + r + lh * 8;
        Out[(size_t)m * D_MODEL + c] = acc[mi][f][r] + bv;
      }
    }
  }
}

// ---------------- flash attention: one wave per 16-query tile ----------------
// Q,K: f16 [b,h,n,64]; Vt: f16 [b,h,64,n]; Ctx: f16 [b,n,768]
__global__ void __launch_bounds__(256) attn_kernel(
    const f16* __restrict__ Q, const f16* __restrict__ K,
    const f16* __restrict__ Vt, f16* __restrict__ Ctx) {
  __shared__ __align__(32) f16 pbuf[8][16 * 32];   // per-wave P tile (16 q x 32 k)

  const int lane = threadIdx.x & 31;
  const int wib  = threadIdx.x >> 5;
  const int w    = blockIdx.x * 8 + wib;           // 0..6143
  const int qt   = w & 63;                         // query tile within (b,h)
  const int bh   = w >> 6;                         // 0..95
  const int bI   = bh / HEADS, hh = bh % HEADS;
  const int lr = lane & 15, lh = lane >> 4;

  const f16* Qb = Q  + (size_t)bh * SEQ * HD;
  const f16* Kb = K  + (size_t)bh * SEQ * HD;
  const f16* Vb = Vt + (size_t)bh * HD * SEQ;

  const int row0 = qt * 16;
  const f16* qrow = Qb + (size_t)(row0 + lr) * HD + lh * 16;
  const v16h a_lo = cat8(*(const v8h*)(qrow),      *(const v8h*)(qrow + 8));  // d 0..31
  const v16h a_hi = cat8(*(const v8h*)(qrow + 32), *(const v8h*)(qrow + 40)); // d 32..63

  float mrun[8], lrun[8];
  v8f o[4] = {};
#pragma unroll
  for (int r = 0; r < 8; ++r) { mrun[r] = -1e30f; lrun[r] = 0.0f; }

  f16* pb = pbuf[wib];
  const float scale = 0.125f;  // 1/sqrt(64)

  for (int it = 0; it < SEQ / 32; ++it) {
    const int k0 = it * 32;
    const f16* kr0 = Kb + (size_t)(k0 + lr) * HD;        // keys k0..k0+15 (col = lr)
    const f16* kr1 = kr0 + 16 * HD;                      // keys k0+16..k0+31
    __builtin_prefetch(kr0 + 32 * HD, 0, 1);             // next iteration's K tile

    v8f z = {};
    v8f s0 = wmma16(a_lo, *(const v16h*)(kr0 + lh * 16), z);
    s0     = wmma16(a_hi, *(const v16h*)(kr0 + 32 + lh * 16), s0);
    v8f s1 = wmma16(a_lo, *(const v16h*)(kr1 + lh * 16), z);
    s1     = wmma16(a_hi, *(const v16h*)(kr1 + 32 + lh * 16), s1);

#pragma unroll
    for (int r = 0; r < 8; ++r) {
      const float x0 = s0[r] * scale, x1 = s1[r] * scale;
      const float mx = halfmax(fmaxf(x0, x1));
      const float nm = fmaxf(mrun[r], mx);
      const float fc = __expf(mrun[r] - nm);
      const float p0 = __expf(x0 - nm);
      const float p1 = __expf(x1 - nm);
      lrun[r] = lrun[r] * fc + halfsum(p0 + p1);
      mrun[r] = nm;
      o[0][r] *= fc; o[1][r] *= fc; o[2][r] *= fc; o[3][r] *= fc;
      const int rr = r + lh * 8;                          // query row in tile
      pb[rr * 32 + lr]      = (f16)p0;                    // C-layout -> LDS row-major
      pb[rr * 32 + 16 + lr] = (f16)p1;
    }
    asm volatile("s_wait_dscnt 0" ::: "memory");          // LDS store->load fence

    const v16h ap = cat8(*(const v8h*)(pb + lr * 32 + lh * 16),
                         *(const v8h*)(pb + lr * 32 + lh * 16 + 8));  // A-layout P
#pragma unroll
    for (int f = 0; f < 4; ++f) {
      const f16* vr = Vb + (size_t)(16 * f + lr) * SEQ + k0 + lh * 16;
      o[f] = wmma16(ap, *(const v16h*)(vr), o[f]);
    }
  }

#pragma unroll
  for (int r = 0; r < 8; ++r) {
    const float inv = 1.0f / lrun[r];
    const int n = row0 + r + lh * 8;
    f16* crow = Ctx + ((size_t)bI * SEQ + n) * D_MODEL + hh * HD;
#pragma unroll
    for (int f = 0; f < 4; ++f)
      crow[16 * f + lr] = (f16)(o[f][r] * inv);
  }
}

// ---------------- host side ----------------
extern "C" void kernel_launch(void* const* d_in, const int* in_sizes, int n_in,
                              void* d_out, int out_size, void* d_ws, size_t ws_size,
                              hipStream_t stream) {
  (void)in_sizes; (void)n_in; (void)out_size; (void)ws_size;
  const float* x  = (const float*)d_in[0];
  const float* Wq = (const float*)d_in[1];
  const float* bq = (const float*)d_in[2];
  const float* Wk = (const float*)d_in[3];
  const float* bk = (const float*)d_in[4];
  const float* Wv = (const float*)d_in[5];
  const float* bv = (const float*)d_in[6];
  const float* Wo = (const float*)d_in[7];
  const float* bo = (const float*)d_in[8];

  char* ws = (char*)d_ws;
  size_t off = 0;
  auto alloc = [&](size_t bytes) -> void* {
    void* p = ws + off;
    off += (bytes + 255) & ~(size_t)255;
    return p;
  };
  f16* xh  = (f16*)alloc((size_t)M_ROWS * D_MODEL * sizeof(f16));
  f16* wqt = (f16*)alloc((size_t)D_MODEL * D_MODEL * sizeof(f16));
  f16* wkt = (f16*)alloc((size_t)D_MODEL * D_MODEL * sizeof(f16));
  f16* wvt = (f16*)alloc((size_t)D_MODEL * D_MODEL * sizeof(f16));
  f16* wot = (f16*)alloc((size_t)D_MODEL * D_MODEL * sizeof(f16));
  f16* Qh  = (f16*)alloc((size_t)M_ROWS * D_MODEL * sizeof(f16));  // [b,h,n,d]
  f16* Kh  = (f16*)alloc((size_t)M_ROWS * D_MODEL * sizeof(f16));  // [b,h,n,d]
  f16* Vth = (f16*)alloc((size_t)M_ROWS * D_MODEL * sizeof(f16));  // [b,h,d,n]
  f16* Ch  = (f16*)alloc((size_t)M_ROWS * D_MODEL * sizeof(f16));  // [b,n,768]

  const int nx = M_ROWS * D_MODEL;
  const int nw = D_MODEL * D_MODEL;
  cvt_f16_kernel<<<(nx + 255) / 256, 256, 0, stream>>>(x, xh, nx);
  cvt_f16_t_kernel<<<(nw + 255) / 256, 256, 0, stream>>>(Wq, wqt, D_MODEL, D_MODEL);
  cvt_f16_t_kernel<<<(nw + 255) / 256, 256, 0, stream>>>(Wk, wkt, D_MODEL, D_MODEL);
  cvt_f16_t_kernel<<<(nw + 255) / 256, 256, 0, stream>>>(Wv, wvt, D_MODEL, D_MODEL);
  cvt_f16_t_kernel<<<(nw + 255) / 256, 256, 0, stream>>>(Wo, wot, D_MODEL, D_MODEL);

  // 32x64 tile per wave: (8192/32)*(768/64) = 3072 waves -> 384 blocks of 8 waves
  const int gemmBlocks = (M_ROWS / 32) * (D_MODEL / 64) / 8;
  proj_gemm_kernel<<<gemmBlocks, 256, 0, stream>>>(xh, wqt, bq, Qh, 0);
  proj_gemm_kernel<<<gemmBlocks, 256, 0, stream>>>(xh, wkt, bk, Kh, 0);
  proj_gemm_kernel<<<gemmBlocks, 256, 0, stream>>>(xh, wvt, bv, Vth, 1);

  const int attnBlocks = BATCH * HEADS * (SEQ / 16) / 8;       // 768 blocks
  attn_kernel<<<attnBlocks, 256, 0, stream>>>(Qh, Kh, Vth, Ch);

  out_gemm_kernel<<<gemmBlocks, 256, 0, stream>>>(Ch, wot, bo, (float*)d_out);
}
